// Net_14267881357843
// MI455X (gfx1250) — matrix operand order repaired and checked by reading, not compiled, not run on hardware
//
#include <hip/hip_runtime.h>
#include <hip/hip_bf16.h>
#include <math.h>

typedef _Float16 f16;
typedef __attribute__((ext_vector_type(16))) _Float16 v16h;
typedef __attribute__((ext_vector_type(8)))  float    v8f;
typedef __attribute__((ext_vector_type(2)))  _Float16 h2;

// ---------------------------------------------------------------------------
// ordered atomic max on float (works for any sign, init must be -inf bits)
// ---------------------------------------------------------------------------
__device__ inline void atomicMaxF(float* addr, float v) {
    if (v >= 0.0f) atomicMax((int*)addr, __float_as_int(v));
    else           atomicMin((unsigned int*)addr, __float_as_uint(v));
}

// ---------------------------------------------------------------------------
// utility kernels
// ---------------------------------------------------------------------------
__global__ void zero_half_kernel(f16* p, long long n) {
    long long i = (long long)blockIdx.x * blockDim.x + threadIdx.x;
    if (i < n) p[i] = (f16)0.0f;
}

__global__ void init_neg_inf_kernel(float* p, int n) {
    int i = blockIdx.x * blockDim.x + threadIdx.x;
    if (i < n) p[i] = -__builtin_inff();
}

// dst[r*ldd + dcol + c] = (f16) src[r*lds + scol + c]
__global__ void cast_copy_kernel(f16* dst, int ldd, int dcol,
                                 const float* src, int lds_, int scol,
                                 int M, int Kc) {
    long long idx = (long long)blockIdx.x * blockDim.x + threadIdx.x;
    if (idx >= (long long)M * Kc) return;
    int r = (int)(idx / Kc), c = (int)(idx % Kc);
    dst[(size_t)r * ldd + dcol + c] = (f16)src[(size_t)r * lds_ + scol + c];
}

// dst[i] = 0.25*(src[i] + src[nb0] + src[nb1] + src[nb2]) row-wise, C columns
__global__ void gather_avg_kernel(const f16* __restrict__ src, f16* __restrict__ dst,
                                  const int* __restrict__ nb, int Npts, int C) {
    long long idx = (long long)blockIdx.x * blockDim.x + threadIdx.x;
    if (idx >= (long long)Npts * C) return;
    int i = (int)(idx / C), c = (int)(idx % C);
    int n0 = nb[i * 3 + 0], n1 = nb[i * 3 + 1], n2 = nb[i * 3 + 2];
    float v = (float)src[(size_t)i  * C + c] + (float)src[(size_t)n0 * C + c] +
              (float)src[(size_t)n1 * C + c] + (float)src[(size_t)n2 * C + c];
    dst[idx] = (f16)(0.25f * v);
}

// ---------------------------------------------------------------------------
// Phase A: spatial MLP + corner conv MLP + kernel-correlation per point.
// Writes spatial[N,64] f16 and structural[N,160] f16 (cols 131..159 zero).
// ---------------------------------------------------------------------------
__global__ void phase_a_kernel(const float* __restrict__ centre,
                               const float* __restrict__ corner,
                               const float* __restrict__ normal,
                               const int*   __restrict__ neighbour,
                               const float* __restrict__ W_sp2, const float* __restrict__ b_sp2,
                               const float* __restrict__ W_sp1, const float* __restrict__ b_sp1,
                               const float* __restrict__ conv_w, const float* __restrict__ conv_b,
                               const float* __restrict__ W_fr3, const float* __restrict__ b_fr3,
                               const float* __restrict__ W_fr4, const float* __restrict__ b_fr4,
                               const float* __restrict__ theta, const float* __restrict__ phi,
                               f16* __restrict__ spatial_out,
                               f16* __restrict__ structural_out,
                               int Npts) {
    __shared__ float sWsp2[64 * 3], sbsp2[64], sWsp1[64 * 64], sbsp1[64];
    __shared__ float sconvw[32 * 6], sconvb[32];
    __shared__ float sWfr3[64 * 32], sbfr3[64], sWfr4[64 * 64], sbfr4[64];
    __shared__ float sKt[64 * 4 * 3];

    for (int x = threadIdx.x; x < 64 * 3;  x += blockDim.x) sWsp2[x] = W_sp2[x];
    for (int x = threadIdx.x; x < 64;      x += blockDim.x) sbsp2[x] = b_sp2[x];
    for (int x = threadIdx.x; x < 64 * 64; x += blockDim.x) sWsp1[x] = W_sp1[x];
    for (int x = threadIdx.x; x < 64;      x += blockDim.x) sbsp1[x] = b_sp1[x];
    for (int x = threadIdx.x; x < 32 * 6;  x += blockDim.x) sconvw[x] = conv_w[x];
    for (int x = threadIdx.x; x < 32;      x += blockDim.x) sconvb[x] = conv_b[x];
    for (int x = threadIdx.x; x < 64 * 32; x += blockDim.x) sWfr3[x] = W_fr3[x];
    for (int x = threadIdx.x; x < 64;      x += blockDim.x) sbfr3[x] = b_fr3[x];
    for (int x = threadIdx.x; x < 64 * 64; x += blockDim.x) sWfr4[x] = W_fr4[x];
    for (int x = threadIdx.x; x < 64;      x += blockDim.x) sbfr4[x] = b_fr4[x];
    for (int x = threadIdx.x; x < 256;     x += blockDim.x) {
        float th = theta[x], ph = phi[x];
        sKt[x * 3 + 0] = sinf(th) * sinf(ph);
        sKt[x * 3 + 1] = sinf(th) * cosf(ph);
        sKt[x * 3 + 2] = cosf(th);
    }
    __syncthreads();

    int i = blockIdx.x * blockDim.x + threadIdx.x;
    if (i >= Npts) return;

    // ---- spatial = lin(relu(lin(centre)))
    float c0 = centre[i * 3 + 0], c1 = centre[i * 3 + 1], c2 = centre[i * 3 + 2];
    float t1[64];
    #pragma unroll 4
    for (int j = 0; j < 64; ++j) {
        float s = sWsp2[j * 3 + 0] * c0 + sWsp2[j * 3 + 1] * c1 + sWsp2[j * 3 + 2] * c2 + sbsp2[j];
        t1[j] = fmaxf(s, 0.0f);
    }
    f16* sp = spatial_out + (size_t)i * 64;
    for (int j2 = 0; j2 < 64; ++j2) {
        float s = sbsp1[j2];
        const float* wr = &sWsp1[j2 * 64];
        #pragma unroll 8
        for (int j = 0; j < 64; ++j) s += t1[j] * wr[j];
        sp[j2] = (f16)s;
    }

    // ---- corner conv + MLP -> frc (structural cols 0..63)
    float x12[12];
    #pragma unroll
    for (int k = 0; k < 9; ++k) x12[k] = corner[i * 9 + k];
    x12[9] = x12[0]; x12[10] = x12[1]; x12[11] = x12[2];
    float comean[32];
    for (int o = 0; o < 32; ++o) {
        float s = 0.0f;
        #pragma unroll
        for (int t = 0; t < 3; ++t)
            #pragma unroll
            for (int k = 0; k < 6; ++k) s += x12[t * 3 + k] * sconvw[o * 6 + k];
        comean[o] = s * (1.0f / 3.0f) + sconvb[o];
    }
    float t3[64];
    for (int j = 0; j < 64; ++j) {
        float s = sbfr3[j];
        const float* wr = &sWfr3[j * 32];
        #pragma unroll 8
        for (int k = 0; k < 32; ++k) s += comean[k] * wr[k];
        t3[j] = fmaxf(s, 0.0f);
    }
    f16* st = structural_out + (size_t)i * 160;
    for (int j2 = 0; j2 < 64; ++j2) {
        float s = sbfr4[j2];
        const float* wr = &sWfr4[j2 * 64];
        #pragma unroll 8
        for (int j = 0; j < 64; ++j) s += t3[j] * wr[j];
        st[j2] = (f16)s;
    }

    // ---- kernel correlation -> structural cols 64..127
    int n0 = neighbour[i * 3 + 0], n1 = neighbour[i * 3 + 1], n2 = neighbour[i * 3 + 2];
    float P[4][3];
    #pragma unroll
    for (int d = 0; d < 3; ++d) {
        P[0][d] = normal[(size_t)i  * 3 + d];
        P[1][d] = normal[(size_t)n0 * 3 + d];
        P[2][d] = normal[(size_t)n1 * 3 + d];
        P[3][d] = normal[(size_t)n2 * 3 + d];
    }
    for (int kk = 0; kk < 64; ++kk) {
        float s = 0.0f;
        #pragma unroll
        for (int q = 0; q < 4; ++q) {
            float kx = sKt[(kk * 4 + q) * 3 + 0];
            float ky = sKt[(kk * 4 + q) * 3 + 1];
            float kz = sKt[(kk * 4 + q) * 3 + 2];
            #pragma unroll
            for (int p = 0; p < 4; ++p) {
                float dx = P[p][0] - kx, dy = P[p][1] - ky, dz = P[p][2] - kz;
                float d2 = dx * dx + dy * dy + dz * dz;
                s += expf(-12.5f * d2);   // 1/(2*sigma^2) = 12.5 for sigma=0.2
            }
        }
        st[64 + kk] = (f16)(s * (1.0f / 16.0f));
    }
    // ---- normal -> cols 128..130, zero pad 131..159
    st[128] = (f16)P[0][0]; st[129] = (f16)P[0][1]; st[130] = (f16)P[0][2];
    for (int c = 131; c < 160; ++c) st[c] = (f16)0.0f;
}

// ---------------------------------------------------------------------------
// WMMA GEMM with async direct-to-LDS staging + double buffering.
// C[N, Mout] = concat(A1[:, :K1], A2[:, :Ktot-K1]) @ W[Mout,Ktot]^T + bias
// 256 threads (8 waves). Block tile: 128 rows x 64 cols. Wave w: rows 16w..16w+15.
// K1 and Ktot must be multiples of 32 (a K-tile never straddles sources).
// OOB rows are clamped to row Nrows-1 (async loads cannot zero-fill); all
// epilogue paths guard row < Nrows so clamped rows are never consumed.
// gmax == nullptr : store f16 to Cout (ldc stride).
// gmax != nullptr : column-max over rows, atomic-max into gmax[ncol].
// ---------------------------------------------------------------------------
__global__ void gemm_wmma_kernel(const f16* __restrict__ A1, int lda1, int K1,
                                 const f16* __restrict__ A2, int lda2,
                                 const f16* __restrict__ W, int Ktot,
                                 const float* __restrict__ bias,
                                 int Nrows,
                                 f16* __restrict__ Cout, int ldc,
                                 float* __restrict__ gmax) {
    __shared__ __align__(16) f16 As[2 * 128 * 32];
    __shared__ __align__(16) f16 Bs[2 * 64 * 32];
    __shared__ float smax[8 * 64];

    const int t    = threadIdx.x;
    const int wv   = t >> 5;
    const int lane = t & 31;
    const int rowBlk = blockIdx.x * 128;
    const int colBlk = blockIdx.y * 64;
    const int nloc = lane & 15;
    const int grp  = lane >> 4;

    v8f acc[4];
    #pragma unroll
    for (int nt = 0; nt < 4; ++nt)
        #pragma unroll
        for (int e = 0; e < 8; ++e) acc[nt][e] = 0.0f;

    // per-thread staging assignments (fixed across tiles)
    const int ar    = t >> 1;             // A row within tile     (0..127)
    const int apart = (t & 1) * 16;       // A half-row (halves)   {0,16}
    int aRow = rowBlk + ar;
    if (aRow >= Nrows) aRow = Nrows - 1;  // clamp (results guarded later)
    const int bm    = t >> 2;             // B (output-col) row    (0..63)
    const int bchnk = (t & 3) * 8;        // B k-chunk (halves)    {0,8,16,24}

    // issue one K-tile's async global->LDS copies (3 x b128 per thread)
    auto issueTile = [&](int k0, int buf) {
        const f16* asrc; int acol;
        if (k0 < K1) { asrc = A1 + (size_t)aRow * lda1; acol = k0 + apart; }
        else         { asrc = A2 + (size_t)aRow * lda2; acol = k0 - K1 + apart; }
        unsigned ldsA = (unsigned)(unsigned long long)(As + buf * (128 * 32) + ar * 32 + apart);
        unsigned long long gA = (unsigned long long)(asrc + acol);
        asm volatile("global_load_async_to_lds_b128 %0, %1, off"
                     :: "v"(ldsA), "v"(gA) : "memory");
        asm volatile("global_load_async_to_lds_b128 %0, %1, off offset:16"
                     :: "v"(ldsA), "v"(gA) : "memory");
        unsigned ldsB = (unsigned)(unsigned long long)(Bs + buf * (64 * 32) + bm * 32 + bchnk);
        unsigned long long gB = (unsigned long long)(W + (size_t)(colBlk + bm) * Ktot + k0 + bchnk);
        asm volatile("global_load_async_to_lds_b128 %0, %1, off"
                     :: "v"(ldsB), "v"(gB) : "memory");
    };

    const int nk = Ktot >> 5;
    issueTile(0, 0);

    for (int kt = 0; kt < nk; ++kt) {
        const int cur = kt & 1;
        if (kt + 1 < nk) {
            issueTile((kt + 1) << 5, cur ^ 1);
            asm volatile("s_wait_asynccnt 3" ::: "memory"); // current tile's 3 done
        } else {
            asm volatile("s_wait_asynccnt 0" ::: "memory");
        }
        __syncthreads();   // all waves' current-tile LDS writes visible

        // ---- A fragment (16-bit A 16x32 layout, ISA 7.12.2)
        v16h a;
        {
            const f16* Arow = As + cur * (128 * 32) + (wv * 16 + nloc) * 32;
            #pragma unroll
            for (int j = 0; j < 8; ++j) {
                int base = (j < 4) ? (grp ? 8 + 2 * j : 2 * j)
                                   : (grp ? 24 + 2 * (j - 4) : 16 + 2 * (j - 4));
                h2 p = *(const h2*)(Arow + base);
                a[2 * j] = p[0]; a[2 * j + 1] = p[1];
            }
        }
        // ---- 4 column tiles
        #pragma unroll
        for (int nt = 0; nt < 4; ++nt) {
            v16h b;
            const f16* Brow = Bs + cur * (64 * 32) + (nt * 16 + nloc) * 32 + grp * 8;
            #pragma unroll
            for (int j = 0; j < 8; ++j) {
                h2 p = *(const h2*)(Brow + ((j < 4) ? 2 * j : 16 + 2 * (j - 4)));
                b[2 * j] = p[0]; b[2 * j + 1] = p[1];
            }
            acc[nt] = __builtin_amdgcn_wmma_f32_16x16x32_f16(
                false, a, false, b, (short)0, acc[nt], false, false);
        }
        __syncthreads();   // done reading `cur` before it is refilled next iter
    }

    // ---- epilogue (C/D layout: lane n = lane&15, row m = r + 8*(lane>>4))
    if (gmax == nullptr) {
        #pragma unroll
        for (int nt = 0; nt < 4; ++nt) {
            int ncol = colBlk + nt * 16 + nloc;
            float bv = bias[ncol];
            #pragma unroll
            for (int r = 0; r < 8; ++r) {
                int row = rowBlk + wv * 16 + r + 8 * grp;
                if (row < Nrows)
                    Cout[(size_t)row * ldc + ncol] = (f16)(acc[nt][r] + bv);
            }
        }
    } else {
        #pragma unroll
        for (int nt = 0; nt < 4; ++nt) {
            int ncol = colBlk + nt * 16 + nloc;
            float bv = bias[ncol];
            float mx = -__builtin_inff();
            #pragma unroll
            for (int r = 0; r < 8; ++r) {
                int row = rowBlk + wv * 16 + r + 8 * grp;
                if (row < Nrows) mx = fmaxf(mx, acc[nt][r] + bv);
            }
            mx = fmaxf(mx, __shfl_xor(mx, 16, 32));
            smax[wv * 64 + nt * 16 + nloc] = mx;
        }
        __syncthreads();
        if (t < 64) {
            float v = -__builtin_inff();
            #pragma unroll
            for (int w = 0; w < 8; ++w) v = fmaxf(v, smax[w * 64 + t]);
            atomicMaxF(&gmax[colBlk + t], v);
        }
    }
}

// ---------------------------------------------------------------------------
// Head: g[1024] -> relu(W31 g) -> relu(W32 h1) -> W33 h2 -> logits[40]
// ---------------------------------------------------------------------------
__global__ void head_kernel(const float* __restrict__ g,
                            const float* __restrict__ W31, const float* __restrict__ b31,
                            const float* __restrict__ W32, const float* __restrict__ b32,
                            const float* __restrict__ W33, const float* __restrict__ b33,
                            float* __restrict__ out) {
    __shared__ float sg[1024], h1[512], h2s[256];
    int t = threadIdx.x;           // blockDim == 512
    for (int j = t; j < 1024; j += 512) sg[j] = g[j];
    __syncthreads();
    {
        float s = b31[t];
        const float* wr = W31 + (size_t)t * 1024;
        for (int k = 0; k < 1024; ++k) s += wr[k] * sg[k];
        h1[t] = fmaxf(s, 0.0f);
    }
    __syncthreads();
    if (t < 256) {
        float s = b32[t];
        const float* wr = W32 + (size_t)t * 512;
        for (int k = 0; k < 512; ++k) s += wr[k] * h1[k];
        h2s[t] = fmaxf(s, 0.0f);
    }
    __syncthreads();
    if (t < 40) {
        float s = b33[t];
        const float* wr = W33 + (size_t)t * 256;
        for (int k = 0; k < 256; ++k) s += wr[k] * h2s[k];
        out[t] = s;
    }
}

// ---------------------------------------------------------------------------
// launch
// ---------------------------------------------------------------------------
static inline int cdiv_i(long long a, long long b) { return (int)((a + b - 1) / b); }

extern "C" void kernel_launch(void* const* d_in, const int* in_sizes, int n_in,
                              void* d_out, int out_size, void* d_ws, size_t ws_size,
                              hipStream_t stream) {
    const int N = in_sizes[0] / 3;

    const float* centre  = (const float*)d_in[0];
    const float* corner  = (const float*)d_in[1];
    const float* normal  = (const float*)d_in[2];
    const int*   nb      = (const int*)  d_in[3];
    const float* W_sp2 = (const float*)d_in[4],  *b_sp2 = (const float*)d_in[5];
    const float* W_sp1 = (const float*)d_in[6],  *b_sp1 = (const float*)d_in[7];
    const float* conv_w= (const float*)d_in[8],  *conv_b= (const float*)d_in[9];
    const float* W_fr3 = (const float*)d_in[10], *b_fr3 = (const float*)d_in[11];
    const float* W_fr4 = (const float*)d_in[12], *b_fr4 = (const float*)d_in[13];
    const float* theta = (const float*)d_in[14], *phi   = (const float*)d_in[15];
    const float* W_c1  = (const float*)d_in[16], *b_c1  = (const float*)d_in[17];
    const float* W_a1  = (const float*)d_in[18], *b_a1  = (const float*)d_in[19];
    const float* W_c2  = (const float*)d_in[20], *b_c2  = (const float*)d_in[21];
    const float* W_a2  = (const float*)d_in[22], *b_a2  = (const float*)d_in[23];
    const float* W_f   = (const float*)d_in[24], *b_f   = (const float*)d_in[25];
    const float* W_m2  = (const float*)d_in[26], *b_m2  = (const float*)d_in[27];
    const float* W_m31 = (const float*)d_in[28], *b_m31 = (const float*)d_in[29];
    const float* W_m32 = (const float*)d_in[30], *b_m32 = (const float*)d_in[31];
    const float* W_m33 = (const float*)d_in[32], *b_m33 = (const float*)d_in[33];

    // workspace carve-up (256B aligned)
    char* ws = (char*)d_ws;
    size_t off = 0;
    auto alloc = [&](size_t bytes) -> void* {
        void* p = ws + off;
        off += (bytes + 255) & ~(size_t)255;
        return p;
    };
    f16* spatial    = (f16*)alloc((size_t)N * 64   * 2);
    f16* structural = (f16*)alloc((size_t)N * 160  * 2);
    f16* agg1       = (f16*)alloc((size_t)N * 160  * 2);
    f16* out2buf    = (f16*)alloc((size_t)N * 256  * 2);
    f16* agg2       = (f16*)alloc((size_t)N * 256  * 2);
    f16* out4buf    = (f16*)alloc((size_t)N * 512  * 2);
    f16* fea        = (f16*)alloc((size_t)N * 1792 * 2);
    f16* Wc1p = (f16*)alloc((size_t)256  * 224  * 2);
    f16* Wa1p = (f16*)alloc((size_t)256  * 160  * 2);
    f16* Wc2p = (f16*)alloc((size_t)512  * 512  * 2);
    f16* Wa2p = (f16*)alloc((size_t)512  * 256  * 2);
    f16* Wfp  = (f16*)alloc((size_t)1024 * 1024 * 2);
    f16* Wm2p = (f16*)alloc((size_t)1024 * 1792 * 2);
    float* g  = (float*)alloc(1024 * 4);

    const int TB = 256;
    // ---- weight casts (with zero padding / concat-boundary remap)
    zero_half_kernel<<<cdiv_i(256LL * 224, TB), TB, 0, stream>>>(Wc1p, 256LL * 224);
    cast_copy_kernel<<<cdiv_i(256LL * 64,  TB), TB, 0, stream>>>(Wc1p, 224, 0,  W_c1, 195, 0,  256, 64);
    cast_copy_kernel<<<cdiv_i(256LL * 131, TB), TB, 0, stream>>>(Wc1p, 224, 64, W_c1, 195, 64, 256, 131);
    zero_half_kernel<<<cdiv_i(256LL * 160, TB), TB, 0, stream>>>(Wa1p, 256LL * 160);
    cast_copy_kernel<<<cdiv_i(256LL * 131, TB), TB, 0, stream>>>(Wa1p, 160, 0, W_a1, 131, 0, 256, 131);
    cast_copy_kernel<<<cdiv_i(512LL * 512,   TB), TB, 0, stream>>>(Wc2p, 512,  0, W_c2, 512,  0, 512,  512);
    cast_copy_kernel<<<cdiv_i(512LL * 256,   TB), TB, 0, stream>>>(Wa2p, 256,  0, W_a2, 256,  0, 512,  256);
    cast_copy_kernel<<<cdiv_i(1024LL * 1024, TB), TB, 0, stream>>>(Wfp,  1024, 0, W_f,  1024, 0, 1024, 1024);
    cast_copy_kernel<<<cdiv_i(1024LL * 1792, TB), TB, 0, stream>>>(Wm2p, 1792, 0, W_m2, 1792, 0, 1024, 1792);

    // ---- phase A
    phase_a_kernel<<<cdiv_i(N, 128), 128, 0, stream>>>(
        centre, corner, normal, nb,
        W_sp2, b_sp2, W_sp1, b_sp1, conv_w, conv_b,
        W_fr3, b_fr3, W_fr4, b_fr4, theta, phi,
        spatial, structural, N);

    // agg1 = neighbour-avg(structural)
    gather_avg_kernel<<<cdiv_i((long long)N * 160, TB), TB, 0, stream>>>(structural, agg1, nb, N, 160);

    const int RB = cdiv_i(N, 128);
    // out1 = [spatial | structural] @ Wc1p^T + b_c1 -> fea[:, 0:256]
    gemm_wmma_kernel<<<dim3(RB, 4), TB, 0, stream>>>(
        spatial, 64, 64, structural, 160, Wc1p, 224, b_c1, N, fea, 1792, nullptr);
    // out2 = agg1 @ Wa1p^T + b_a1 -> out2buf
    gemm_wmma_kernel<<<dim3(RB, 4), TB, 0, stream>>>(
        agg1, 160, 160, nullptr, 0, Wa1p, 160, b_a1, N, out2buf, 256, nullptr);
    // agg2 = neighbour-avg(out2)
    gather_avg_kernel<<<cdiv_i((long long)N * 256, TB), TB, 0, stream>>>(out2buf, agg2, nb, N, 256);
    // out3 = [out1 | out2] @ Wc2p^T + b_c2 -> fea[:, 256:768]
    gemm_wmma_kernel<<<dim3(RB, 8), TB, 0, stream>>>(
        fea, 1792, 256, out2buf, 256, Wc2p, 512, b_c2, N, fea + 256, 1792, nullptr);
    // out4 = agg2 @ Wa2p^T + b_a2 -> out4buf
    gemm_wmma_kernel<<<dim3(RB, 8), TB, 0, stream>>>(
        agg2, 256, 256, nullptr, 0, Wa2p, 256, b_a2, N, out4buf, 512, nullptr);
    // fused = [out3 | out4] @ Wfp^T + b_f -> fea[:, 768:1792]
    gemm_wmma_kernel<<<dim3(RB, 16), TB, 0, stream>>>(
        fea + 256, 1792, 512, out4buf, 512, Wfp, 1024, b_f, N, fea + 768, 1792, nullptr);

    // g = max_rows(fea @ Wm2p^T + b_m2)
    init_neg_inf_kernel<<<4, TB, 0, stream>>>(g, 1024);
    gemm_wmma_kernel<<<dim3(RB, 16), TB, 0, stream>>>(
        fea, 1792, 1792, nullptr, 0, Wm2p, 1792, b_m2, N, nullptr, 0, g);

    // head
    head_kernel<<<1, 512, 0, stream>>>(g, W_m31, b_m31, W_m32, b_m32, W_m33, b_m33, (float*)d_out);
}
